// Traj_Model_8461085573280
// MI455X (gfx1250) — matrix operand order repaired and checked by reading, not compiled, not run on hardware
//
#include <hip/hip_runtime.h>
#include <hip/hip_bf16.h>

// ---------------------------------------------------------------------------
// MoE (8 experts, top-2) for MI455X / gfx1250, wave32, bf16 WMMA.
// Memory-bound problem (268 MB fp32 weights @ 23.3 TB/s ~= 11.5us floor), so
// we stream weights once per token tile, run dense over experts (gate==0
// cancels non-top-2 contributions -> deterministic, no atomics), and keep all
// matrix math on v_wmma_f32_16x16x32_bf16.
//
// Round-1 fix: inner GEMM uses 4 independent WMMA accumulator chains (one per
// K-step in a staging phase) so consecutive v_wmma ops have no RAW hazard and
// the compiler doesn't need the 4x v_nop padding seen in round 0's assembly
// (ISA 7.12.1: bf16 WMMA->WMMA RAW needs 5 slots).
// ---------------------------------------------------------------------------

typedef __bf16 v16bf __attribute__((ext_vector_type(16)));
typedef float  v8f   __attribute__((ext_vector_type(8)));
typedef unsigned int v8u __attribute__((ext_vector_type(8)));
typedef float  v4f   __attribute__((ext_vector_type(4)));

#define TOKENS   8192
#define NEMBD    1024
#define HID      4096
#define NEXP     8
#define MTILE    64      // tokens per workgroup
#define HCHUNK   128     // hidden cols per fused chunk

__device__ __forceinline__ unsigned short f2bf(float f) {
  __bf16 h = (__bf16)f;
  return __builtin_bit_cast(unsigned short, h);
}
__device__ __forceinline__ unsigned packbf2(float a, float b) {
  return (unsigned)f2bf(a) | ((unsigned)f2bf(b) << 16);
}
__device__ __forceinline__ float gelu_tanh(float t) {
  return 0.5f * t * (1.0f + tanhf(0.7978845608028654f * (t + 0.044715f * t * t * t)));
}

// ---------------------------------------------------------------------------
// Kernel 1: gating. One wave32 per token. Writes dense softmax (gate1) to the
// tail of d_out and the renormalized top-2 sparse gates [8192][8] to ws.
// ---------------------------------------------------------------------------
__global__ __launch_bounds__(256) void moe_gate_kernel(
    const float* __restrict__ x, const float* __restrict__ dt,
    const float* __restrict__ dd, const float* __restrict__ drg,
    const float* __restrict__ de, const float* __restrict__ cityemb,
    const float* __restrict__ Wg, const float* __restrict__ bg,
    const int* __restrict__ city_index,
    float* __restrict__ gate1_out, float* __restrict__ gates_ws) {
  const int wave = (blockIdx.x * blockDim.x + threadIdx.x) >> 5;
  const int lane = threadIdx.x & 31;
  if (wave >= TOKENS) return;

  const float* city = cityemb + (*city_index) * 32;
  const float* xs  = x   + (size_t)wave * NEMBD;
  const float* dts = dt  + (size_t)wave * 256;
  const float* dds = dd  + (size_t)wave * 256;
  const float* drs = drg + (size_t)wave * 128;
  const float* des = de  + (size_t)wave * 128;

  float acc[NEXP];
#pragma unroll
  for (int e = 0; e < NEXP; ++e) acc[e] = 0.0f;

  for (int d = lane; d < 1824; d += 32) {
    float f;
    if (d < 1024)      f = xs[d];
    else if (d < 1056) f = city[d - 1024];
    else if (d < 1312) f = dts[d - 1056];
    else if (d < 1568) f = dds[d - 1312];
    else if (d < 1696) f = drs[d - 1568];
    else               f = des[d - 1696];
    const float* wr = Wg + (size_t)d * NEXP;
#pragma unroll
    for (int e = 0; e < NEXP; ++e) acc[e] += f * wr[e];
  }
#pragma unroll
  for (int e = 0; e < NEXP; ++e) {
#pragma unroll
    for (int off = 16; off > 0; off >>= 1) acc[e] += __shfl_xor(acc[e], off, 32);
  }
  if (lane == 0) {
    float lg[NEXP], ex[NEXP];
    float mx = -1e30f;
#pragma unroll
    for (int e = 0; e < NEXP; ++e) { lg[e] = acc[e] + bg[e]; mx = fmaxf(mx, lg[e]); }
    float se = 0.0f;
#pragma unroll
    for (int e = 0; e < NEXP; ++e) { ex[e] = expf(lg[e] - mx); se += ex[e]; }
    const float inv = 1.0f / se;
#pragma unroll
    for (int e = 0; e < NEXP; ++e) gate1_out[(size_t)wave * NEXP + e] = ex[e] * inv;
    // top-2 (first index wins ties, like lax.top_k)
    int i0 = 0;
#pragma unroll
    for (int e = 1; e < NEXP; ++e) if (lg[e] > lg[i0]) i0 = e;
    int i1 = (i0 == 0) ? 1 : 0;
#pragma unroll
    for (int e = 0; e < NEXP; ++e) if (e != i0 && lg[e] > lg[i1]) i1 = e;
    const float g0 = 1.0f / (1.0f + expf(lg[i1] - lg[i0]));
    float gz[NEXP];
#pragma unroll
    for (int e = 0; e < NEXP; ++e) gz[e] = 0.0f;
    gz[i0] = g0; gz[i1] = 1.0f - g0;
#pragma unroll
    for (int e = 0; e < NEXP; ++e) gates_ws[(size_t)wave * NEXP + e] = gz[e];
  }
}

// ---------------------------------------------------------------------------
// Kernel 2: fused MoE FFN. 64-token tile per workgroup, dense over experts,
// gate folded into h so out accumulates directly (no atomics, deterministic).
// LDS plan (dynamic, 242 KB of the 320 KB/WGP):
//   x_u   [64][512]  u32 (bf16 pairs, k-major)        131072 B
//   h_s   [64][128]  bf16 (gelu(xW1+b1)*gate)          16384 B
//   w1s   [128 cols][128 k]  bf16 transposed           32768 B
//   w2s   [1024 cols][32 k]  bf16 transposed           65536 B
//   g_s   [64][8]    f32 gates                          2048 B
// ---------------------------------------------------------------------------
__global__ __launch_bounds__(1024) void moe_ffn_kernel(
    const float* __restrict__ x,
    const float* __restrict__ W1, const float* __restrict__ b1,
    const float* __restrict__ W2, const float* __restrict__ b2,
    const float* __restrict__ gates_ws, float* __restrict__ out) {
  extern __shared__ char smem[];
  unsigned*       x_u = (unsigned*)smem;                                   // 131072
  unsigned short* h_s = (unsigned short*)(smem + 131072);                  //  16384
  unsigned short* w1s = (unsigned short*)(smem + 131072 + 16384);          //  32768
  unsigned short* w2s = (unsigned short*)(smem + 131072 + 16384 + 32768);  //  65536
  float*          g_s = (float*)(smem + 131072 + 16384 + 32768 + 65536);   //   2048

  const int tid  = threadIdx.x;
  const int wave = tid >> 5;
  const int lane = tid & 31;
  const int half = lane >> 4;   // lane group (ISA 7.12.2 layouts)
  const int ln   = lane & 15;
  const int tok0 = blockIdx.x * MTILE;

  // ---- stage x tile as bf16 (packed u32 pairs), coalesced b128 reads ----
  {
    const int m  = tid >> 4;          // 0..63
    const int kb = (tid & 15) * 64;   // 0..960
    const v4f* s4 = (const v4f*)(x + (size_t)(tok0 + m) * NEMBD + kb);
    unsigned* dst = x_u + m * 512 + (kb >> 1);
#pragma unroll
    for (int j = 0; j < 16; ++j) {
      v4f v = s4[j];
      dst[2 * j]     = packbf2(v.x, v.y);
      dst[2 * j + 1] = packbf2(v.z, v.w);
    }
  }
  if (tid < MTILE * NEXP) g_s[tid] = gates_ws[(size_t)tok0 * NEXP + tid];
  __syncthreads();

  // output accumulator: wave owns out[0:64, wave*32 : wave*32+32]
  v8f oa[4][2];
#pragma unroll
  for (int tm = 0; tm < 4; ++tm)
#pragma unroll
    for (int tn = 0; tn < 2; ++tn) oa[tm][tn] = (v8f)0.0f;

  const int mi = wave & 3;   // inner-gemm tile: rows mi*16
  const int ni = wave >> 2;  // inner-gemm tile: h-cols ni*16 (of 128)

  for (int e = 0; e < NEXP; ++e) {
    const size_t wbase = (size_t)e * NEMBD * HID;  // same for W1 & W2
    for (int hc = 0; hc < HID / HCHUNK; ++hc) {
      // ================= inner GEMM: h = x @ W1[:, hc*128..) =================
      // 4 independent accumulator chains -> the 4 WMMAs inside a staging phase
      // have no mutual RAW hazard (no v_nop padding); chains are summed in a
      // fixed order below (deterministic).
      v8f hp[4];
#pragma unroll
      for (int s = 0; s < 4; ++s) hp[s] = (v8f)0.0f;
#pragma unroll 1
      for (int p = 0; p < 8; ++p) {  // K slices of 128
        {  // stage W1 slice [128 k][128 h] -> w1s[h][k] bf16 (transposed)
          const int kk = tid >> 3;
          const int hb = (tid & 7) * 16;
          const float* src = W1 + wbase + (size_t)(p * 128 + kk) * HID + hc * HCHUNK + hb;
          __builtin_prefetch(src + (size_t)128 * HID, 0, 0);  // next K slice
          const v4f* s4 = (const v4f*)src;
#pragma unroll
          for (int j = 0; j < 4; ++j) {
            v4f v = s4[j];
            w1s[(hb + 4 * j + 0) * 128 + kk] = f2bf(v.x);
            w1s[(hb + 4 * j + 1) * 128 + kk] = f2bf(v.y);
            w1s[(hb + 4 * j + 2) * 128 + kk] = f2bf(v.z);
            w1s[(hb + 4 * j + 3) * 128 + kk] = f2bf(v.w);
          }
        }
        __syncthreads();
#pragma unroll
        for (int s = 0; s < 4; ++s) {  // K steps of 32, independent chains
          const int kb = p * 128 + s * 32;
          v8u au, bu;
#pragma unroll
          for (int v = 0; v < 8; ++v) {  // A layout: K = base + half*8 + (v>>2)*16 + (v&3)*2
            const int k = kb + half * 8 + (v >> 2) * 16 + (v & 3) * 2;
            au[v] = x_u[(mi * 16 + ln) * 512 + (k >> 1)];
          }
#pragma unroll
          for (int v = 0; v < 8; ++v) {  // B layout: K = half*8 + (v>>2)*16 + (v&3)*2 (local)
            const int kl = s * 32 + half * 8 + (v >> 2) * 16 + (v & 3) * 2;
            bu[v] = *(const unsigned*)&w1s[(ni * 16 + ln) * 128 + kl];
          }
          hp[s] = __builtin_amdgcn_wmma_f32_16x16x32_bf16(
              false, __builtin_bit_cast(v16bf, au),
              false, __builtin_bit_cast(v16bf, bu),
              (short)0, hp[s], false, false);
        }
        __syncthreads();  // WAR on w1s before restage
      }
      // gelu(h + b1) * gate  -> bf16 h_s   (gate folded here: gemm2 can
      // accumulate straight into the final output accumulator)
      {
        const int hcol = ni * 16 + ln;
        const float b1v = b1[(size_t)e * HID + hc * HCHUNK + hcol];
#pragma unroll
        for (int r = 0; r < 8; ++r) {
          const int m = mi * 16 + r + half * 8;  // C/D layout: M = r + 8*half
          const float gv = g_s[m * NEXP + e];
          const float hsum = ((hp[0][r] + hp[1][r]) + (hp[2][r] + hp[3][r]));
          h_s[m * 128 + hcol] = f2bf(gelu_tanh(hsum + b1v) * gv);
        }
      }
      __syncthreads();

      // ================= outer GEMM: out += h @ W2[hc*128.., :] =============
#pragma unroll 1
      for (int q = 0; q < 4; ++q) {  // K2 slices of 32
        {  // stage W2 slice [32 k][1024 c] -> w2s[c][k] bf16 (transposed)
          const int r  = tid >> 5;
          const int cb = (tid & 31) * 32;
          const float* src = W2 + wbase + (size_t)(hc * HCHUNK + q * 32 + r) * NEMBD + cb;
          __builtin_prefetch(src + (size_t)32 * NEMBD, 0, 0);
          const v4f* s4 = (const v4f*)src;
#pragma unroll
          for (int j = 0; j < 8; ++j) {
            v4f v = s4[j];
            w2s[(cb + 4 * j + 0) * 32 + r] = f2bf(v.x);
            w2s[(cb + 4 * j + 1) * 32 + r] = f2bf(v.y);
            w2s[(cb + 4 * j + 2) * 32 + r] = f2bf(v.z);
            w2s[(cb + 4 * j + 3) * 32 + r] = f2bf(v.w);
          }
        }
        __syncthreads();
        // B frags depend only on tn -> build once, reuse across 4 tm tiles.
        // The 8 WMMAs below hit 8 distinct accumulators -> no RAW padding.
        v8u bu[2];
#pragma unroll
        for (int tn = 0; tn < 2; ++tn) {
          const int col = wave * 32 + tn * 16 + ln;
#pragma unroll
          for (int v = 0; v < 8; ++v) {
            const int kl = half * 8 + (v >> 2) * 16 + (v & 3) * 2;
            bu[tn][v] = *(const unsigned*)&w2s[col * 32 + kl];
          }
        }
#pragma unroll
        for (int tm = 0; tm < 4; ++tm) {
          v8u au;
#pragma unroll
          for (int v = 0; v < 8; ++v) {
            const int k = q * 32 + half * 8 + (v >> 2) * 16 + (v & 3) * 2;
            au[v] = *(const unsigned*)&h_s[(tm * 16 + ln) * 128 + k];
          }
#pragma unroll
          for (int tn = 0; tn < 2; ++tn) {
            oa[tm][tn] = __builtin_amdgcn_wmma_f32_16x16x32_bf16(
                false, __builtin_bit_cast(v16bf, au),
                false, __builtin_bit_cast(v16bf, bu[tn]),
                (short)0, oa[tm][tn], false, false);
          }
        }
        __syncthreads();  // WAR on w2s
      }
    }
  }

  // ---- epilogue: + sum_e gate[m][e]*b2[e][col], coalesced store ----
  float b2c[2][NEXP];
#pragma unroll
  for (int tn = 0; tn < 2; ++tn) {
    const int col = wave * 32 + tn * 16 + ln;
#pragma unroll
    for (int e = 0; e < NEXP; ++e) b2c[tn][e] = b2[(size_t)e * NEMBD + col];
  }
#pragma unroll
  for (int tm = 0; tm < 4; ++tm) {
#pragma unroll
    for (int r = 0; r < 8; ++r) {
      const int m = tm * 16 + r + half * 8;
      float gr[NEXP];
#pragma unroll
      for (int e = 0; e < NEXP; ++e) gr[e] = g_s[m * NEXP + e];
#pragma unroll
      for (int tn = 0; tn < 2; ++tn) {
        const int col = wave * 32 + tn * 16 + ln;
        float v = oa[tm][tn][r];
#pragma unroll
        for (int e = 0; e < NEXP; ++e) v += gr[e] * b2c[tn][e];
        out[(size_t)(tok0 + m) * NEMBD + col] = v;
      }
    }
  }
}

// ---------------------------------------------------------------------------
extern "C" void kernel_launch(void* const* d_in, const int* in_sizes, int n_in,
                              void* d_out, int out_size, void* d_ws, size_t ws_size,
                              hipStream_t stream) {
  const float* x    = (const float*)d_in[0];
  const float* dt   = (const float*)d_in[1];
  const float* dd   = (const float*)d_in[2];
  const float* drg  = (const float*)d_in[3];
  const float* de   = (const float*)d_in[4];
  const float* city = (const float*)d_in[5];
  const float* Wg   = (const float*)d_in[6];
  const float* bg   = (const float*)d_in[7];
  const float* W1   = (const float*)d_in[8];
  const float* b1   = (const float*)d_in[9];
  const float* W2   = (const float*)d_in[10];
  const float* b2   = (const float*)d_in[11];
  const int* cidx   = (const int*)d_in[12];

  float* out_main  = (float*)d_out;                       // [4,2048,1024]
  float* out_gate1 = out_main + (size_t)TOKENS * NEMBD;   // [4,2048,8]
  float* gates_ws  = (float*)d_ws;                        // [8192][8] sparse gates

  // gating: one wave32/token, 8 waves/block
  moe_gate_kernel<<<TOKENS / 8, 256, 0, stream>>>(
      x, dt, dd, drg, de, city, Wg, bg, cidx, out_gate1, gates_ws);

  // fused MoE FFN: 128 WGs x 1024 threads, 242 KB dynamic LDS
  const size_t smem = 131072 + 16384 + 32768 + 65536 + 2048;
  moe_ffn_kernel<<<TOKENS / MTILE, 1024, smem, stream>>>(
      x, W1, b1, W2, b2, gates_ws, out_main);
}
